// DynamicConvolution_21878563406172
// MI455X (gfx1250) — matrix operand order repaired and checked by reading, not compiled, not run on hardware
//
#include <hip/hip_runtime.h>
#include <hip/hip_bf16.h>

// ---------------------------------------------------------------------------
// DynamicConvolution for MI455X (gfx1250, wave32, WMMA bf16 + async->LDS)
//   T=4096, B=8, C=512, H=8, K=7  ->  M = T*B = 32768 flattened rows
// ---------------------------------------------------------------------------

typedef __bf16 bf16;
typedef __attribute__((ext_vector_type(16))) __bf16 v16bf;
typedef __attribute__((ext_vector_type(8)))  __bf16 v8bf;
typedef __attribute__((ext_vector_type(8)))  float  v8f;
typedef __attribute__((ext_vector_type(4)))  float  v4f;

#define T_DIM 4096
#define B_DIM 8
#define C_DIM 512
#define M_DIM (T_DIM * B_DIM)   /* 32768 */
#define HK_DIM 56               /* H*K */
#define NWT_PAD 64              /* HK padded to multiple of 16 */

static __device__ __forceinline__ v8f wmma_bf16(v16bf a, v16bf b, v8f c) {
  // 8-arg form: (neg_a, A, neg_b, B, c_mod, C, reuse_a, reuse_b)
  return __builtin_amdgcn_wmma_f32_16x16x32_bf16(false, a, false, b, (short)0, c,
                                                 false, false);
}

// Async copy of one 16-byte chunk global -> LDS (ASYNCcnt-tracked).
static __device__ __forceinline__ void async_copy_16B(unsigned lds_addr,
                                                      const void* gptr) {
  asm volatile("global_load_async_to_lds_b128 %0, %1, off"
               :: "v"(lds_addr), "v"(gptr) : "memory");
}

static __device__ __forceinline__ void async_wait0() {
  asm volatile("s_wait_asynccnt 0x0" ::: "memory");
}

// Low 32 bits of an LDS generic pointer = LDS byte offset (ISA 10.2 aperture).
static __device__ __forceinline__ unsigned lds_off(const void* p) {
  return (unsigned)(uintptr_t)p;
}

// ---------------------------------------------------------------------------
// Kernel 1: convert weights f32 -> bf16 (w_wt padded to 64 rows with zeros)
// ---------------------------------------------------------------------------
__global__ void dc_cvt_weights(const float* __restrict__ w_in,
                               const float* __restrict__ w_wt,
                               const float* __restrict__ w_out,
                               bf16* __restrict__ w_in_bf,
                               bf16* __restrict__ w_wt_bf,
                               bf16* __restrict__ w_out_bf) {
  const int N_IN  = 2 * C_DIM * C_DIM;       // 1024*512
  const int N_WT  = NWT_PAD * C_DIM;         // 64*512 (padded)
  const int N_OUT = C_DIM * C_DIM;           // 512*512
  int idx = blockIdx.x * blockDim.x + threadIdx.x;
  if (idx < N_IN) {
    w_in_bf[idx] = (bf16)w_in[idx];
  } else if (idx < N_IN + N_WT) {
    int j = idx - N_IN;
    int r = j >> 9;                          // row (0..63)
    w_wt_bf[j] = (r < HK_DIM) ? (bf16)w_wt[j] : (bf16)0.0f;
  } else if (idx < N_IN + N_WT + N_OUT) {
    int j = idx - N_IN - N_WT;
    w_out_bf[j] = (bf16)w_out[j];
  }
}

// ---------------------------------------------------------------------------
// Kernel 2: h = x @ w_in^T + b_in ; GLU: xg = a * sigmoid(g)  (bf16 out)
//   grid (M/16, 4), block 256 (8 waves); wave w -> 16 xg columns.
//   A tile (16x512 f32) async-staged in LDS once, shared by all 8 waves.
//   Two WMMA accumulators per wave (columns n and n+512) share one A frag.
// ---------------------------------------------------------------------------
__global__ void dc_gemm_glu(const float* __restrict__ x,
                            const bf16* __restrict__ w_in_bf,
                            const float* __restrict__ b_in,
                            bf16* __restrict__ xg) {
  __shared__ float lds_x[16 * C_DIM];                  // 32 KB
  const int lane  = threadIdx.x & 31;
  const int wave  = threadIdx.x >> 5;
  const int m0    = blockIdx.x * 16;
  const int n0    = blockIdx.y * 128 + wave * 16;      // [0,512)
  const int lhalf = (lane >> 4) & 1;
  const int nc    = n0 + (lane & 15);

  // Stage contiguous 32 KB A tile: 2048 x 16B chunks over 256 threads.
  {
    const unsigned lbase = lds_off(&lds_x[0]);
    const char* gsrc = (const char*)(x + (size_t)m0 * C_DIM);
#pragma unroll
    for (int i = 0; i < 8; ++i) {
      const int chunk = i * 256 + (int)threadIdx.x;
      async_copy_16B(lbase + chunk * 16, gsrc + (size_t)chunk * 16);
    }
    async_wait0();
    __syncthreads();
  }

  const float* xrow = lds_x + (lane & 15) * C_DIM;
  const bf16*  wa   = w_in_bf + (size_t)nc * C_DIM + lhalf * 8;
  const bf16*  wg   = w_in_bf + (size_t)(C_DIM + nc) * C_DIM + lhalf * 8;

  v8f acc_a = {0.f, 0.f, 0.f, 0.f, 0.f, 0.f, 0.f, 0.f};
  v8f acc_g = {0.f, 0.f, 0.f, 0.f, 0.f, 0.f, 0.f, 0.f};

  for (int k0 = 0; k0 < C_DIM; k0 += 32) {
    __builtin_prefetch(wa + k0 + 64, 0, 3);
    __builtin_prefetch(wg + k0 + 64, 0, 3);
    const int kbase = k0 + lhalf * 8;
    // A fragment from LDS: rows m=lane&15, K runs [kbase..+7], [kbase+16..+23]
    v4f r0 = *(const v4f*)(xrow + kbase);
    v4f r1 = *(const v4f*)(xrow + kbase + 4);
    v4f r2 = *(const v4f*)(xrow + kbase + 16);
    v4f r3 = *(const v4f*)(xrow + kbase + 20);
    v16bf afrag;
#pragma unroll
    for (int e = 0; e < 4; ++e) {
      afrag[e]      = (bf16)r0[e];
      afrag[e + 4]  = (bf16)r1[e];
      afrag[e + 8]  = (bf16)r2[e];
      afrag[e + 12] = (bf16)r3[e];
    }
    // B fragments: column nc, K runs [k0+8*lhalf..+7], [k0+16+8*lhalf..+7]
    v8bf ba0 = *(const v8bf*)(wa + k0);
    v8bf ba1 = *(const v8bf*)(wa + k0 + 16);
    v8bf bg0 = *(const v8bf*)(wg + k0);
    v8bf bg1 = *(const v8bf*)(wg + k0 + 16);
    v16bf bfA, bfG;
#pragma unroll
    for (int e = 0; e < 8; ++e) {
      bfA[e] = ba0[e]; bfA[e + 8] = ba1[e];
      bfG[e] = bg0[e]; bfG[e + 8] = bg1[e];
    }
    acc_a = wmma_bf16(afrag, bfA, acc_a);
    acc_g = wmma_bf16(afrag, bfG, acc_g);
  }

  const float biasA = b_in[nc];
  const float biasG = b_in[C_DIM + nc];
#pragma unroll
  for (int v = 0; v < 8; ++v) {
    const int m  = m0 + v + lhalf * 8;       // C/D row layout
    const float aa = acc_a[v] + biasA;
    const float gg = acc_g[v] + biasG;
    const float val = aa / (1.0f + __expf(-gg));
    xg[(size_t)m * C_DIM + nc] = (bf16)val;
  }
}

// ---------------------------------------------------------------------------
// Kernel 3: logits = xg @ w_wt^T + b_wt -> softmax over K=7 taps per head.
//   grid (M/16), block 128 (4 waves); wave w -> columns [16w,16w+16) of 64.
//   A tile (16x512 bf16) async-staged in LDS once.
// ---------------------------------------------------------------------------
__global__ void dc_dynw(const bf16* __restrict__ xg,
                        const bf16* __restrict__ w_wt_bf,
                        const float* __restrict__ b_wt,
                        float* __restrict__ dynw) {
  __shared__ bf16  lds_a[16 * C_DIM];                  // 16 KB
  __shared__ float logits[16][NWT_PAD];
  const int lane  = threadIdx.x & 31;
  const int wave  = threadIdx.x >> 5;                  // 0..3
  const int m0    = blockIdx.x * 16;
  const int n0    = wave * 16;
  const int lhalf = (lane >> 4) & 1;
  const int nc    = n0 + (lane & 15);

  // Stage contiguous 16 KB A tile: 1024 x 16B chunks over 128 threads.
  {
    const unsigned lbase = lds_off(&lds_a[0]);
    const char* gsrc = (const char*)(xg + (size_t)m0 * C_DIM);
#pragma unroll
    for (int i = 0; i < 8; ++i) {
      const int chunk = i * 128 + (int)threadIdx.x;
      async_copy_16B(lbase + chunk * 16, gsrc + (size_t)chunk * 16);
    }
    async_wait0();
    __syncthreads();
  }

  const bf16* arow = lds_a + (lane & 15) * C_DIM;
  const bf16* wr   = w_wt_bf + (size_t)nc * C_DIM + lhalf * 8;

  v8f acc = {0.f, 0.f, 0.f, 0.f, 0.f, 0.f, 0.f, 0.f};
  for (int k0 = 0; k0 < C_DIM; k0 += 32) {
    __builtin_prefetch(wr + k0 + 64, 0, 3);
    const int kbase = k0 + lhalf * 8;
    v8bf a0 = *(const v8bf*)(arow + kbase);
    v8bf a1 = *(const v8bf*)(arow + kbase + 16);
    v8bf b0 = *(const v8bf*)(wr + k0);
    v8bf b1 = *(const v8bf*)(wr + k0 + 16);
    v16bf af, bfW;
#pragma unroll
    for (int e = 0; e < 8; ++e) {
      af[e]  = a0[e]; af[e + 8]  = a1[e];
      bfW[e] = b0[e]; bfW[e + 8] = b1[e];
    }
    acc = wmma_bf16(af, bfW, acc);
  }

  const float bias = (nc < HK_DIM) ? b_wt[nc] : 0.0f;
#pragma unroll
  for (int v = 0; v < 8; ++v)
    logits[v + lhalf * 8][nc] = acc[v] + bias;
  __syncthreads();

  // softmax over 7 taps: 128 threads -> (m, h)
  const int tid = threadIdx.x;
  const int m = tid >> 3;
  const int h = tid & 7;
  float vals[7];
  float mx = -3.0e38f;
#pragma unroll
  for (int k = 0; k < 7; ++k) {
    vals[k] = logits[m][h * 7 + k];
    mx = fmaxf(mx, vals[k]);
  }
  float s = 0.0f;
#pragma unroll
  for (int k = 0; k < 7; ++k) { vals[k] = __expf(vals[k] - mx); s += vals[k]; }
  const float inv = 1.0f / s;
#pragma unroll
  for (int k = 0; k < 7; ++k)
    dynw[(size_t)(m0 + m) * HK_DIM + h * 7 + k] = vals[k] * inv;
}

// ---------------------------------------------------------------------------
// Kernel 4: depthwise dynamic conv over time (PAD_L = 3), + conv_bias.
//   out[m][c] = sum_k xg[m + (k-3)*B][c] * dynw[m][c/64][k]   (bf16 out)
// ---------------------------------------------------------------------------
__global__ void dc_conv(const bf16* __restrict__ xg,
                        const float* __restrict__ dynw,
                        const float* __restrict__ conv_bias,
                        bf16* __restrict__ convo) {
  const int idx = blockIdx.x * blockDim.x + threadIdx.x;   // < M*C
  const int m = idx >> 9;
  const int c = idx & (C_DIM - 1);
  const int h = c >> 6;                                    // C/H = 64
  const float* wr = dynw + (size_t)m * HK_DIM + h * 7;
  float acc = conv_bias[c];
#pragma unroll
  for (int k = 0; k < 7; ++k) {
    const int row = m + (k - 3) * B_DIM;                   // time shift
    if (row >= 0 && row < M_DIM)
      acc += (float)xg[(size_t)row * C_DIM + c] * wr[k];
  }
  convo[idx] = (bf16)acc;
}

// ---------------------------------------------------------------------------
// Kernel 5: y = conv @ w_out^T + b_out   (f32 out to d_out)
//   A tile (16x512 bf16) async-staged in LDS once, shared by all 8 waves.
// ---------------------------------------------------------------------------
__global__ void dc_gemm_out(const bf16* __restrict__ convo,
                            const bf16* __restrict__ w_out_bf,
                            const float* __restrict__ b_out,
                            float* __restrict__ y) {
  __shared__ bf16 lds_a[16 * C_DIM];                   // 16 KB
  const int lane  = threadIdx.x & 31;
  const int wave  = threadIdx.x >> 5;
  const int m0    = blockIdx.x * 16;
  const int n0    = blockIdx.y * 128 + wave * 16;
  const int lhalf = (lane >> 4) & 1;
  const int nc    = n0 + (lane & 15);

  // Stage contiguous 16 KB A tile: 1024 x 16B chunks over 256 threads.
  {
    const unsigned lbase = lds_off(&lds_a[0]);
    const char* gsrc = (const char*)(convo + (size_t)m0 * C_DIM);
#pragma unroll
    for (int i = 0; i < 4; ++i) {
      const int chunk = i * 256 + (int)threadIdx.x;
      async_copy_16B(lbase + chunk * 16, gsrc + (size_t)chunk * 16);
    }
    async_wait0();
    __syncthreads();
  }

  const bf16* arow = lds_a + (lane & 15) * C_DIM;
  const bf16* wr   = w_out_bf + (size_t)nc * C_DIM + lhalf * 8;

  v8f acc = {0.f, 0.f, 0.f, 0.f, 0.f, 0.f, 0.f, 0.f};
  for (int k0 = 0; k0 < C_DIM; k0 += 32) {
    __builtin_prefetch(wr + k0 + 64, 0, 3);
    const int kbase = k0 + lhalf * 8;
    v8bf a0 = *(const v8bf*)(arow + kbase);
    v8bf a1 = *(const v8bf*)(arow + kbase + 16);
    v8bf b0 = *(const v8bf*)(wr + k0);
    v8bf b1 = *(const v8bf*)(wr + k0 + 16);
    v16bf af, bfW;
#pragma unroll
    for (int e = 0; e < 8; ++e) {
      af[e]  = a0[e]; af[e + 8]  = a1[e];
      bfW[e] = b0[e]; bfW[e + 8] = b1[e];
    }
    acc = wmma_bf16(af, bfW, acc);
  }

  const float bias = b_out[nc];
#pragma unroll
  for (int v = 0; v < 8; ++v) {
    const int m = m0 + v + lhalf * 8;
    y[(size_t)m * C_DIM + nc] = acc[v] + bias;
  }
}

// ---------------------------------------------------------------------------
// Host-side launcher
// ---------------------------------------------------------------------------
extern "C" void kernel_launch(void* const* d_in, const int* in_sizes, int n_in,
                              void* d_out, int out_size, void* d_ws, size_t ws_size,
                              hipStream_t stream) {
  const float* x         = (const float*)d_in[0];
  const float* w_in      = (const float*)d_in[1];
  const float* b_in      = (const float*)d_in[2];
  const float* w_wt      = (const float*)d_in[3];
  const float* b_wt      = (const float*)d_in[4];
  const float* w_out     = (const float*)d_in[5];
  const float* b_out     = (const float*)d_in[6];
  const float* conv_bias = (const float*)d_in[7];
  float* y = (float*)d_out;

  // Workspace partition (all region sizes are multiples of 256 bytes).
  char* ws = (char*)d_ws;
  bf16* w_in_bf  = (bf16*)ws;  ws += (size_t)2 * C_DIM * C_DIM * sizeof(bf16);
  bf16* w_wt_bf  = (bf16*)ws;  ws += (size_t)NWT_PAD * C_DIM * sizeof(bf16);
  bf16* w_out_bf = (bf16*)ws;  ws += (size_t)C_DIM * C_DIM * sizeof(bf16);
  bf16* xg       = (bf16*)ws;  ws += (size_t)M_DIM * C_DIM * sizeof(bf16);
  float* dynw    = (float*)ws; ws += (size_t)M_DIM * HK_DIM * sizeof(float);
  bf16* convo    = (bf16*)ws;

  const int n_cvt = 2 * C_DIM * C_DIM + NWT_PAD * C_DIM + C_DIM * C_DIM;
  dc_cvt_weights<<<(n_cvt + 255) / 256, 256, 0, stream>>>(
      w_in, w_wt, w_out, w_in_bf, w_wt_bf, w_out_bf);

  dc_gemm_glu<<<dim3(M_DIM / 16, 4), 256, 0, stream>>>(x, w_in_bf, b_in, xg);

  dc_dynw<<<M_DIM / 16, 128, 0, stream>>>(xg, w_wt_bf, b_wt, dynw);

  dc_conv<<<(M_DIM * C_DIM) / 256, 256, 0, stream>>>(xg, dynw, conv_bias, convo);

  dc_gemm_out<<<dim3(M_DIM / 16, 4), 256, 0, stream>>>(convo, w_out_bf, b_out, y);
}
